// GeometricUnit_43611097923824
// MI455X (gfx1250) — matrix operand order repaired and checked by reading, not compiled, not run on hardware
//
#include <hip/hip_runtime.h>
#include <math.h>

typedef __attribute__((ext_vector_type(16))) __bf16 v16bf;
typedef __attribute__((ext_vector_type(8)))  float  v8f;
typedef __attribute__((ext_vector_type(4)))  unsigned int v4u;

#define L_SEQ 512
#define BT    128
#define IN_K  512
#define H_DIM 60

// ---------------------------------------------------------------------------
// Kernel 1: logits = inp @ W.T (+b), softmax over H, phi = atan2(x@sinA, x@cosA)
// bf16 WMMA 16x16x32. Block = 128 threads (4 waves), each wave owns 16 rows,
// block owns 64 rows of M = L*B = 65536.  W staged once per block in LDS (bf16,
// padded stride 520 -> conflict-free ds_load_b128 B-fragments; row 60 = zeros
// serves as the N-padding for n in [60,64)).
// ---------------------------------------------------------------------------
struct B32frag { v4u lo; v4u hi; };

__launch_bounds__(128)
__global__ void k_gemm_softmax_phi(const float* __restrict__ inp,
                                   const float* __restrict__ W,
                                   const float* __restrict__ bias,
                                   const float* __restrict__ alphabet,
                                   float* __restrict__ phi_out)   // (L*3, 128)
{
    // Phase A: W bf16, 61 rows x stride 520 = 63440 B.
    // Phase B (aliased after W is dead): logits 64x65 f32 (16640 B) + sin/cos alphabet (1440 B).
    __shared__ __align__(16) char smem[61 * 520 * 2];
    __bf16* Wl = (__bf16*)smem;
    float*  Ls = (float*)smem;                  // [64][65]
    float*  sA = (float*)(smem + 64 * 65 * 4);  // [60][3]
    float*  cA = sA + 180;

    const int tid  = threadIdx.x;
    const int lane = tid & 31;
    const int wv   = tid >> 5;
    const int ln15 = lane & 15;
    const int lh   = lane >> 4;          // 0 or 1 (lane half)

    // ---- stage W -> LDS as bf16 (row 60 zeroed for N padding) ----
    for (int idx = tid; idx < 61 * 512; idx += 128) {
        int r = idx >> 9, c = idx & 511;
        float v = (r < H_DIM) ? W[r * IN_K + c] : 0.0f;
        Wl[r * 520 + c] = (__bf16)v;
    }
    __syncthreads();

    const int    mrow = blockIdx.x * 64 + wv * 16 + ln15;
    const float* arow = inp + (size_t)mrow * IN_K;
    const int    h8   = lh * 8;

    v8f acc[4] = {};   // 4 N-tiles x (16x16 f32 / wave)

    for (int ks = 0; ks < 16; ++ks) {          // K = 512 = 16 * 32
        // A fragment: lane holds row m=ln15; K = ks*32 + 16*lh + [0..15] (contiguous)
        const float* pa = arow + ks * 32 + lh * 16;
        v8f f0 = *(const v8f*)(pa);
        v8f f1 = *(const v8f*)(pa + 8);
        v16bf A;
#pragma unroll
        for (int e = 0; e < 8; ++e) A[e]     = (__bf16)f0[e];
#pragma unroll
        for (int e = 0; e < 8; ++e) A[8 + e] = (__bf16)f1[e];

#pragma unroll
        for (int nt = 0; nt < 4; ++nt) {
            // B fragment: lane holds col n; K = ks*32 + 16*lh + [0..15] (contiguous)
            int n = nt * 16 + ln15;
            if (n >= H_DIM) n = H_DIM;          // zero pad row
            const __bf16* pb = Wl + n * 520 + ks * 32 + lh * 16;
            B32frag raw;
            raw.lo = *(const v4u*)(pb);
            raw.hi = *(const v4u*)(pb + 8);
            v16bf Bf = __builtin_bit_cast(v16bf, raw);
            acc[nt] = __builtin_amdgcn_wmma_f32_16x16x32_bf16(
                false, A, false, Bf, (short)0, acc[nt], false, false);
        }
    }
    __syncthreads();   // W region dead; safe to alias

    // stage sin/cos(alphabet)
    for (int idx = tid; idx < 180; idx += 128) {
        float a = alphabet[idx];
        sA[idx] = sinf(a);
        cA[idx] = cosf(a);
    }
    // spill logits: C/D layout -> row = wv*16 + r + 8*lh, col = nt*16 + ln15
#pragma unroll
    for (int nt = 0; nt < 4; ++nt) {
        int col = nt * 16 + ln15;
#pragma unroll
        for (int r = 0; r < 8; ++r)
            Ls[(wv * 16 + r + 8 * lh) * 65 + col] = acc[nt][r];
    }
    __syncthreads();

    if (tid < 64) {
        const float* lr = Ls + tid * 65;
        float mx = -1e30f;
        for (int h = 0; h < H_DIM; ++h) mx = fmaxf(mx, lr[h] + bias[h]);
        float ss0 = 0, ss1 = 0, ss2 = 0, sc0 = 0, sc1 = 0, sc2 = 0;
        for (int h = 0; h < H_DIM; ++h) {
            float e = __expf(lr[h] + bias[h] - mx);
            ss0 += e * sA[h * 3 + 0]; sc0 += e * cA[h * 3 + 0];
            ss1 += e * sA[h * 3 + 1]; sc1 += e * cA[h * 3 + 1];
            ss2 += e * sA[h * 3 + 2]; sc2 += e * cA[h * 3 + 2];
        }
        // softmax 1/sum cancels inside atan2
        int m = blockIdx.x * 64 + tid;
        int l = m >> 7, bch = m & 127;
        float* po = phi_out + (size_t)(l * 3) * BT + bch;
        po[0 * BT] = atan2f(ss0, sc0);
        po[1 * BT] = atan2f(ss1, sc1);
        po[2 * BT] = atan2f(ss2, sc2);
    }
}

// ---------------------------------------------------------------------------
// Kernel 2: parallel affine prefix scan of the NeRF extension.
// Step i: C' = C + R*t_i ; R' = R * M(phi_i), with
//   M = [ ct  -st    0 ]      t = r * (ct, cp*st, sp*st)
//       [cp*st cp*ct -sp]
//       [sp*st sp*ct  cp]
// One block per batch element; 256 threads x 6 steps.
// ---------------------------------------------------------------------------
struct Xf { float R[9]; float t[3]; };

__device__ __forceinline__ Xf composeX(const Xf& p, const Xf& q) {
    Xf o;
#pragma unroll
    for (int i = 0; i < 3; ++i) {
#pragma unroll
        for (int j = 0; j < 3; ++j)
            o.R[i * 3 + j] = p.R[i * 3 + 0] * q.R[0 + j]
                           + p.R[i * 3 + 1] * q.R[3 + j]
                           + p.R[i * 3 + 2] * q.R[6 + j];
        o.t[i] = p.t[i] + p.R[i * 3 + 0] * q.t[0]
                        + p.R[i * 3 + 1] * q.t[1]
                        + p.R[i * 3 + 2] * q.t[2];
    }
    return o;
}

__device__ __forceinline__ Xf makeT(float sp, float cp, float st, float ct, float r) {
    Xf T;
    T.R[0] = ct;      T.R[1] = -st;     T.R[2] = 0.f;
    T.R[3] = cp * st; T.R[4] = cp * ct; T.R[5] = -sp;
    T.R[6] = sp * st; T.R[7] = sp * ct; T.R[8] = cp;
    T.t[0] = r * ct;  T.t[1] = r * cp * st; T.t[2] = r * sp * st;
    return T;
}

__device__ __forceinline__ void storeX(float* s, int tid, const Xf& x) {
    float* p = s + tid * 12;
#pragma unroll
    for (int j = 0; j < 9; ++j) p[j] = x.R[j];
    p[9] = x.t[0]; p[10] = x.t[1]; p[11] = x.t[2];
}
__device__ __forceinline__ Xf loadX(const float* s, int tid) {
    const float* p = s + tid * 12;
    Xf x;
#pragma unroll
    for (int j = 0; j < 9; ++j) x.R[j] = p[j];
    x.t[0] = p[9]; x.t[1] = p[10]; x.t[2] = p[11];
    return x;
}

__launch_bounds__(256)
__global__ void k_scan(const float* __restrict__ phi,   // (3L, 128)
                       const float* __restrict__ blen,  // (3,)
                       const float* __restrict__ bang,  // (3,)
                       float* __restrict__ out)         // (3L, 128, 3)
{
    __shared__ float sx[256 * 12];
    const int b   = blockIdx.x;
    const int tid = threadIdx.x;

    float r3[3], st3[3], ct3[3];
#pragma unroll
    for (int c = 0; c < 3; ++c) {
        r3[c] = blen[c];
        float a = bang[c];
        st3[c] = sinf(a); ct3[c] = cosf(a);
    }

    // local composition of 6 steps (cache sin/cos(phi) for the replay)
    const int i0 = tid * 6;
    float sphi[6], cphi[6];
    Xf loc = { {1,0,0, 0,1,0, 0,0,1}, {0,0,0} };
#pragma unroll
    for (int s = 0; s < 6; ++s) {
        int c = s % 3;                       // i0 is a multiple of 3
        float p = phi[(size_t)(i0 + s) * BT + b];
        float sp, cp;
        sincosf(p, &sp, &cp);
        sphi[s] = sp; cphi[s] = cp;
        loc = composeX(loc, makeT(sp, cp, st3[c], ct3[c], r3[c]));
    }

    // Hillis-Steele inclusive scan over 256 non-commutative affine transforms
    storeX(sx, tid, loc);
    for (int off = 1; off < 256; off <<= 1) {
        __syncthreads();
        Xf other;
        bool act = (tid >= off);
        if (act) other = loadX(sx, tid - off);
        __syncthreads();
        if (act) { loc = composeX(other, loc); storeX(sx, tid, loc); }
    }
    __syncthreads();

    // initial frame from carry0 = (e0, e1, e2): R0 = [bc0, n0 x bc0, n0], C0 = e2
    const float s2 = 0.70710678f, s3c = 0.57735027f, s6 = 0.40824829f;
    const Xf F0 = { { 0.f,     2.f * s6, s3c,
                     -s2,     -s6,       s3c,
                      s2,     -s6,       s3c }, { 0.f, 0.f, 1.f } };
    Xf start;
    if (tid == 0) start = F0;
    else          start = composeX(F0, loadX(sx, tid - 1));

    // replay 6 steps, emitting coordinates
    float R[9], C[3];
#pragma unroll
    for (int j = 0; j < 9; ++j) R[j] = start.R[j];
#pragma unroll
    for (int j = 0; j < 3; ++j) C[j] = start.t[j];

#pragma unroll
    for (int s = 0; s < 6; ++s) {
        int c = s % 3;
        Xf T = makeT(sphi[s], cphi[s], st3[c], ct3[c], r3[c]);
        float D0 = C[0] + R[0] * T.t[0] + R[1] * T.t[1] + R[2] * T.t[2];
        float D1 = C[1] + R[3] * T.t[0] + R[4] * T.t[1] + R[5] * T.t[2];
        float D2 = C[2] + R[6] * T.t[0] + R[7] * T.t[1] + R[8] * T.t[2];
        float* po = out + ((size_t)(i0 + s) * BT + b) * 3;
        po[0] = D0; po[1] = D1; po[2] = D2;
        float Rn[9];
#pragma unroll
        for (int i = 0; i < 3; ++i)
#pragma unroll
            for (int j = 0; j < 3; ++j)
                Rn[i * 3 + j] = R[i * 3 + 0] * T.R[0 + j]
                              + R[i * 3 + 1] * T.R[3 + j]
                              + R[i * 3 + 2] * T.R[6 + j];
#pragma unroll
        for (int j = 0; j < 9; ++j) R[j] = Rn[j];
        C[0] = D0; C[1] = D1; C[2] = D2;
    }
}

// ---------------------------------------------------------------------------
extern "C" void kernel_launch(void* const* d_in, const int* in_sizes, int n_in,
                              void* d_out, int out_size, void* d_ws, size_t ws_size,
                              hipStream_t stream) {
    const float* inp      = (const float*)d_in[0];   // (512,128,512)
    const float* W        = (const float*)d_in[1];   // (60,512)
    const float* bias     = (const float*)d_in[2];   // (60,)
    const float* alphabet = (const float*)d_in[3];   // (60,3)
    const float* blen     = (const float*)d_in[4];   // (3,)
    const float* bang     = (const float*)d_in[5];   // (3,)
    float* out = (float*)d_out;                      // (1536,128,3)
    float* phi = (float*)d_ws;                       // (1536,128) scratch

    k_gemm_softmax_phi<<<(L_SEQ * BT) / 64, 128, 0, stream>>>(inp, W, bias, alphabet, phi);
    k_scan<<<BT, 256, 0, stream>>>(phi, blen, bang, out);
}
